// InvariantUpdateLayer_36893769072773
// MI455X (gfx1250) — matrix operand order repaired and checked by reading, not compiled, not run on hardware
//
#include <hip/hip_runtime.h>
#include <hip/hip_bf16.h>

#define LN_EPS 1e-5f

typedef __attribute__((ext_vector_type(16))) __bf16 v16bf;
typedef __attribute__((ext_vector_type(8)))  float  v8f;

// ---------- helpers ----------

__device__ __forceinline__ unsigned short f2bf(float f) {
  union { float f; unsigned u; } v; v.f = f;
  unsigned r = v.u + 0x7FFFu + ((v.u >> 16) & 1u);   // round-to-nearest-even
  return (unsigned short)(r >> 16);
}

__device__ __forceinline__ float silu_f(float x) {
  return x / (1.f + __expf(-x));
}

// A fragment (16x32 bf16, MxK) from row-major LDS tile.
// ISA 7.12.2: lanes 0-15 -> row=lane, elems 0..7 = K k0..k0+7, elems 8..15 = K k0+16..k0+23
//             lanes 16-31 -> row=lane-16, elems 0..7 = K k0+8.., elems 8..15 = K k0+24..
__device__ __forceinline__ v16bf ldA(const unsigned short* s, int lda, int k0, int lane) {
  int r = lane & 15;
  int koff = (lane < 16) ? 0 : 8;
  const unsigned short* p = s + r * lda + k0 + koff;
  union { uint4 u[2]; v16bf v; } c;
  c.u[0] = *(const uint4*)p;        // K = k0+koff .. +7
  c.u[1] = *(const uint4*)(p + 16); // K = k0+koff+16 .. +23
  return c.v;
}

// B fragment from pre-repacked weights: 32B contiguous per lane per (kt,nt) tile.
__device__ __forceinline__ v16bf ldB(const unsigned short* Wf, int kt, int nt, int lane) {
  const unsigned short* p = Wf + ((((size_t)kt * 16 + nt) * 32 + lane) << 4);
  union { uint4 u[2]; v16bf v; } c;
  c.u[0] = ((const uint4*)p)[0];
  c.u[1] = ((const uint4*)p)[1];
  return c.v;
}

#define WMMA_BF16(a, b, c) \
  __builtin_amdgcn_wmma_f32_16x16x32_bf16(false, (a), false, (b), (short)0, (c), false, false)

__device__ __forceinline__ void atomicAddF(float* p, float v) {
  __hip_atomic_fetch_add(p, v, __ATOMIC_RELAXED, __HIP_MEMORY_SCOPE_AGENT);
}

// ---------- prep kernels ----------

__global__ void prep_h_agg(const float* __restrict__ h, unsigned short* __restrict__ hb,
                           float* __restrict__ agg, int n) {
  int i = blockIdx.x * blockDim.x + threadIdx.x;
  if (i < n) { agg[i] = 0.f; hb[i] = f2bf(h[i]); }
}

// Repack row-major W[kdim][256] (f32) into per-lane WMMA B-fragment layout (bf16).
// dst index = ((kt*16 + nt)*32 + lane)*16 + i ; src K = kt*32 + (lane<16?0:8) + (i<8?i:i+8), N = nt*16 + (lane&15)
__global__ void repack_w(const float* __restrict__ W, unsigned short* __restrict__ Wf, int kdim) {
  int tid = blockIdx.x * blockDim.x + threadIdx.x;
  if (tid >= kdim * 256) return;
  int i  = tid & 15;
  int l  = (tid >> 4) & 31;
  int nt = (tid >> 9) & 15;
  int kt = tid >> 13;
  int K = kt * 32 + ((l < 16) ? 0 : 8) + ((i < 8) ? i : i + 8);
  int N = nt * 16 + (l & 15);
  Wf[tid] = f2bf(W[(size_t)K * 256 + N]);
}

// ---------- edge kernel: MLP1 + LN + MLP2 + scatter-add ----------

__global__ __launch_bounds__(256) void edge_msg_kernel(
    const int* __restrict__ e, const float* __restrict__ d2,
    const float* __restrict__ mW1_last,   // f32 row 512 of mW1 (the d2 feature)
    const float* __restrict__ mb1, const float* __restrict__ ln_g,
    const float* __restrict__ ln_b, const float* __restrict__ mb2,
    const unsigned short* __restrict__ hb,
    const unsigned short* __restrict__ W1f, const unsigned short* __restrict__ W2f,
    float* __restrict__ agg, int E)
{
  __shared__ unsigned short sA[16 * 512];     // gathered [hi|hj] bf16, 16 KB
  __shared__ float          sM[16 * 260];     // post-SiLU f32 stage (padded rows)
  __shared__ unsigned short sN[16 * 264];     // post-LN bf16 A-tile for GEMM2
  __shared__ float sSum[16][17], sSq[16][17];
  __shared__ float sMu[16], sRs[16];
  __shared__ int   sEi[16], sEj[16];
  __shared__ float sD2[16];

  const int t = threadIdx.x;
  const int ebase = blockIdx.x * 16;
  if (t < 16) {
    sEi[t] = e[ebase + t];
    sEj[t] = e[E + ebase + t];
    sD2[t] = d2[ebase + t];
  }
  __syncthreads();

  { // gather 16 rows of hi|hj (bf16) into LDS: 64B per thread
    int row = t >> 4, seg = t & 15;
    const unsigned short* src = (seg < 8)
        ? (hb + (size_t)sEi[row] * 256 + seg * 32)
        : (hb + (size_t)sEj[row] * 256 + (seg - 8) * 32);
    uint4* d = (uint4*)(sA + row * 512 + seg * 32);
    const uint4* s = (const uint4*)src;
    d[0] = s[0]; d[1] = s[1]; d[2] = s[2]; d[3] = s[3];
  }
  __syncthreads();

  const int w = t >> 5, lane = t & 31;
  const int nt0 = w * 2;                      // each wave owns 2 N-tiles (16x32 output)
  const int colBase = nt0 * 16 + (lane & 15);
  const int rbase = (lane < 16) ? 0 : 8;

  // ---- GEMM1: [16x512] x [512x32] (+ d2 rank-1 + bias), SiLU ----
  v8f acc0 = {}; v8f acc1 = {};
#pragma unroll
  for (int kt = 0; kt < 16; ++kt) {
    v16bf a  = ldA(sA, 512, kt * 32, lane);
    v16bf b0 = ldB(W1f, kt, nt0, lane);
    v16bf b1 = ldB(W1f, kt, nt0 + 1, lane);
    acc0 = WMMA_BF16(a, b0, acc0);
    acc1 = WMMA_BF16(a, b1, acc1);
  }
  {
    float wl0 = mW1_last[colBase], wl1 = mW1_last[colBase + 16];
    float bb0 = mb1[colBase],      bb1 = mb1[colBase + 16];
#pragma unroll
    for (int r = 0; r < 8; ++r) {
      float dd = sD2[rbase + r];
      sM[(rbase + r) * 260 + colBase]      = silu_f(acc0[r] + dd * wl0 + bb0);
      sM[(rbase + r) * 260 + colBase + 16] = silu_f(acc1[r] + dd * wl1 + bb1);
    }
  }
  __syncthreads();

  // ---- LayerNorm over 256 cols per edge (all cols live in this block) ----
  {
    int row = t >> 4, seg = t & 15;
    float s = 0.f, q = 0.f;
#pragma unroll
    for (int i = 0; i < 16; ++i) {
      float x = sM[row * 260 + seg * 16 + i];
      s += x; q += x * x;
    }
    sSum[row][seg] = s; sSq[row][seg] = q;
  }
  __syncthreads();
  if (t < 16) {
    float s = 0.f, q = 0.f;
#pragma unroll
    for (int i = 0; i < 16; ++i) { s += sSum[t][i]; q += sSq[t][i]; }
    float mu  = s * (1.f / 256.f);
    float var = q * (1.f / 256.f) - mu * mu;
    sMu[t] = mu;
    sRs[t] = rsqrtf(var + LN_EPS);
  }
  __syncthreads();
  {
    int row = t >> 4, seg = t & 15;
    float mu = sMu[row], rs = sRs[row];
#pragma unroll
    for (int i = 0; i < 16; ++i) {
      int col = seg * 16 + i;
      float x = sM[row * 260 + col];
      sN[row * 264 + col] = f2bf((x - mu) * rs * ln_g[col] + ln_b[col]);
    }
  }
  __syncthreads();

  // ---- GEMM2: [16x256] x [256x32] + bias, SiLU, scatter-add ----
  v8f c0 = {}; v8f c1 = {};
#pragma unroll
  for (int kt = 0; kt < 8; ++kt) {
    v16bf a  = ldA(sN, 264, kt * 32, lane);
    v16bf b0 = ldB(W2f, kt, nt0, lane);
    v16bf b1 = ldB(W2f, kt, nt0 + 1, lane);
    c0 = WMMA_BF16(a, b0, c0);
    c1 = WMMA_BF16(a, b1, c1);
  }
  {
    float bb0 = mb2[colBase], bb1 = mb2[colBase + 16];
#pragma unroll
    for (int r = 0; r < 8; ++r) {
      float* dst = agg + (size_t)sEi[rbase + r] * 256;
      atomicAddF(dst + colBase,      silu_f(c0[r] + bb0));
      atomicAddF(dst + colBase + 16, silu_f(c1[r] + bb1));
    }
  }
}

// ---------- node kernel: update MLP + residual ----------

__global__ __launch_bounds__(256) void node_update_kernel(
    const float* __restrict__ h, const float* __restrict__ agg,
    const unsigned short* __restrict__ hb,
    const float* __restrict__ ub1, const float* __restrict__ ub2,
    const unsigned short* __restrict__ U1f, const unsigned short* __restrict__ U2f,
    float* __restrict__ out)
{
  __shared__ unsigned short sA[16 * 512];     // [h | agg] bf16
  __shared__ unsigned short sN[16 * 264];

  const int t = threadIdx.x;
  const int nbase = blockIdx.x * 16;

  {
    int row = t >> 4, seg = t & 15;
    size_t node = (size_t)(nbase + row);
    if (seg < 8) {
      const uint4* s = (const uint4*)(hb + node * 256 + seg * 32);
      uint4* d = (uint4*)(sA + row * 512 + seg * 32);
      d[0] = s[0]; d[1] = s[1]; d[2] = s[2]; d[3] = s[3];
    } else {
      const float* s = agg + node * 256 + (seg - 8) * 32;
      unsigned short* d = sA + row * 512 + 256 + (seg - 8) * 32;
#pragma unroll
      for (int i = 0; i < 32; ++i) d[i] = f2bf(s[i]);
    }
  }
  __syncthreads();

  const int w = t >> 5, lane = t & 31;
  const int nt0 = w * 2;
  const int colBase = nt0 * 16 + (lane & 15);
  const int rbase = (lane < 16) ? 0 : 8;

  v8f acc0 = {}; v8f acc1 = {};
#pragma unroll
  for (int kt = 0; kt < 16; ++kt) {
    v16bf a  = ldA(sA, 512, kt * 32, lane);
    v16bf b0 = ldB(U1f, kt, nt0, lane);
    v16bf b1 = ldB(U1f, kt, nt0 + 1, lane);
    acc0 = WMMA_BF16(a, b0, acc0);
    acc1 = WMMA_BF16(a, b1, acc1);
  }
  {
    float bb0 = ub1[colBase], bb1 = ub1[colBase + 16];
#pragma unroll
    for (int r = 0; r < 8; ++r) {
      sN[(rbase + r) * 264 + colBase]      = f2bf(silu_f(acc0[r] + bb0));
      sN[(rbase + r) * 264 + colBase + 16] = f2bf(silu_f(acc1[r] + bb1));
    }
  }
  __syncthreads();

  v8f c0 = {}; v8f c1 = {};
#pragma unroll
  for (int kt = 0; kt < 8; ++kt) {
    v16bf a  = ldA(sN, 264, kt * 32, lane);
    v16bf b0 = ldB(U2f, kt, nt0, lane);
    v16bf b1 = ldB(U2f, kt, nt0 + 1, lane);
    c0 = WMMA_BF16(a, b0, c0);
    c1 = WMMA_BF16(a, b1, c1);
  }
  {
    float bb0 = ub2[colBase], bb1 = ub2[colBase + 16];
#pragma unroll
    for (int r = 0; r < 8; ++r) {
      size_t o0 = (size_t)(nbase + rbase + r) * 256 + colBase;
      out[o0]      = h[o0]      + c0[r] + bb0;
      out[o0 + 16] = h[o0 + 16] + c1[r] + bb1;
    }
  }
}

// ---------- launch ----------

extern "C" void kernel_launch(void* const* d_in, const int* in_sizes, int n_in,
                              void* d_out, int out_size, void* d_ws, size_t ws_size,
                              hipStream_t stream) {
  (void)n_in; (void)out_size; (void)ws_size;
  const float* h   = (const float*)d_in[0];
  const int*   e   = (const int*)d_in[1];
  const float* d2  = (const float*)d_in[2];
  const float* mW1 = (const float*)d_in[3];
  const float* mb1 = (const float*)d_in[4];
  const float* lng = (const float*)d_in[5];
  const float* lnb = (const float*)d_in[6];
  const float* mW2 = (const float*)d_in[7];
  const float* mb2 = (const float*)d_in[8];
  const float* uW1 = (const float*)d_in[9];
  const float* ub1 = (const float*)d_in[10];
  const float* uW2 = (const float*)d_in[11];
  const float* ub2 = (const float*)d_in[12];

  const int N = in_sizes[0] / 256;
  const int E = in_sizes[1] / 2;

  char* ws = (char*)d_ws;
  size_t off = 0;
  float*          agg = (float*)(ws + off);          off += (size_t)N * 256 * sizeof(float);
  unsigned short* hb  = (unsigned short*)(ws + off); off += (size_t)N * 256 * 2;
  unsigned short* W1f = (unsigned short*)(ws + off); off += (size_t)512 * 256 * 2;
  unsigned short* W2f = (unsigned short*)(ws + off); off += (size_t)256 * 256 * 2;
  unsigned short* U1f = (unsigned short*)(ws + off); off += (size_t)512 * 256 * 2;
  unsigned short* U2f = (unsigned short*)(ws + off); off += (size_t)256 * 256 * 2;

  const int nElems = N * 256;
  prep_h_agg<<<(nElems + 255) / 256, 256, 0, stream>>>(h, hb, agg, nElems);
  repack_w<<<(512 * 256 + 255) / 256, 256, 0, stream>>>(mW1, W1f, 512);
  repack_w<<<(256 * 256 + 255) / 256, 256, 0, stream>>>(mW2, W2f, 256);
  repack_w<<<(512 * 256 + 255) / 256, 256, 0, stream>>>(uW1, U1f, 512);
  repack_w<<<(256 * 256 + 255) / 256, 256, 0, stream>>>(uW2, U2f, 256);

  edge_msg_kernel<<<E / 16, 256, 0, stream>>>(e, d2, mW1 + (size_t)512 * 256, mb1,
                                              lng, lnb, mb2, hb, W1f, W2f, agg, E);
  node_update_kernel<<<N / 16, 256, 0, stream>>>(h, agg, hb, ub1, ub2, U1f, U2f,
                                                 (float*)d_out);
}
